// LlamaModel_15925738733624
// MI455X (gfx1250) — matrix operand order repaired and checked by reading, not compiled, not run on hardware
//
#include <hip/hip_runtime.h>
#include <math.h>

typedef __attribute__((ext_vector_type(16))) __bf16 v16bf;
typedef __attribute__((ext_vector_type(8)))  __bf16 v8bf;
typedef __attribute__((ext_vector_type(8)))  float  v8f;

// Problem constants (match reference)
#define B_   2
#define S_   1024
#define D_   2048
#define HQ_  32
#define HKV_ 8
#define HD_  64
#define I_   5632
#define M_   (B_ * S_)            // 2048 tokens
#define NQKV_ ((HQ_ + 2 * HKV_) * HD_)  // 3072
#define EPS_  1e-5f
#define SCALE_ 0.125f             // HD^-0.5
#define LN10000_OVER32 0.28782313662425574f  // ln(10000)/32

// ---------------------------------------------------------------------------
// CDNA5 async global->LDS copy (ASYNCcnt path), inline asm for portability
// across ROCm 7.2 / amdgpu-toolchain builtin arity differences.
// ---------------------------------------------------------------------------
#if defined(__AMDGCN__)
__device__ __forceinline__ unsigned lds_addr_of(const void* p) {
  typedef __attribute__((address_space(3))) const void lds_cv;
  return (unsigned)(unsigned long long)(lds_cv*)p;
}
__device__ __forceinline__ void async_copy_b128_to_lds(const void* gptr,
                                                       const void* lptr) {
  unsigned long long ga = (unsigned long long)gptr;
  unsigned la = lds_addr_of(lptr);
  asm volatile("global_load_async_to_lds_b128 %0, %1, off"
               :: "v"(la), "v"(ga)
               : "memory");
}
__device__ __forceinline__ void wait_asynccnt0() {
  asm volatile("s_wait_asynccnt 0" ::: "memory");
}
#endif

// ---------------------------------------------------------------------------
// Embedding gather + residual zero-init
// ---------------------------------------------------------------------------
__global__ __launch_bounds__(256)
void embed_kernel(const int* __restrict__ ids, const float* __restrict__ embed,
                  float* __restrict__ h, float* __restrict__ res) {
  size_t idx = (size_t)blockIdx.x * 256 + threadIdx.x;   // over M_*D_
  int t = (int)(idx >> 11);      // D_ = 2048 = 2^11
  int c = (int)(idx & (D_ - 1));
  int id = ids[t];
  h[idx]   = embed[(size_t)id * D_ + c];
  res[idx] = 0.0f;
}

// ---------------------------------------------------------------------------
// Fused residual-add + RMSNorm; res <- h + res; out_bf <- bf16(rmsnorm(res)*w)
// ---------------------------------------------------------------------------
__global__ __launch_bounds__(256)
void addnorm_kernel(const float* __restrict__ hin, float* __restrict__ res,
                    const float* __restrict__ w, __bf16* __restrict__ outbf) {
  int row = blockIdx.x;
  int tid = threadIdx.x;
  const float* hr = hin + (size_t)row * D_;
  float*       rr = res + (size_t)row * D_;
  float vals[8];
  float ss = 0.0f;
#pragma unroll
  for (int i = 0; i < 8; ++i) {
    int c = tid + i * 256;
    float v = hr[c] + rr[c];
    vals[i] = v;
    ss += v * v;
  }
  __shared__ float red[256];
  red[tid] = ss;
  __syncthreads();
  for (int off = 128; off > 0; off >>= 1) {
    if (tid < off) red[tid] += red[tid + off];
    __syncthreads();
  }
  float rs = rsqrtf(red[0] * (1.0f / (float)D_) + EPS_);
#pragma unroll
  for (int i = 0; i < 8; ++i) {
    int c = tid + i * 256;
    rr[c] = vals[i];
    outbf[(size_t)row * D_ + c] = (__bf16)(vals[i] * rs * w[c]);
  }
}

// Final RMSNorm: out <- rmsnorm(h+res)*w (fp32)
__global__ __launch_bounds__(256)
void finalnorm_kernel(const float* __restrict__ hin, const float* __restrict__ res,
                      const float* __restrict__ w, float* __restrict__ out) {
  int row = blockIdx.x;
  int tid = threadIdx.x;
  const float* hr = hin + (size_t)row * D_;
  const float* rr = res + (size_t)row * D_;
  float vals[8];
  float ss = 0.0f;
#pragma unroll
  for (int i = 0; i < 8; ++i) {
    int c = tid + i * 256;
    float v = hr[c] + rr[c];
    vals[i] = v;
    ss += v * v;
  }
  __shared__ float red[256];
  red[tid] = ss;
  __syncthreads();
  for (int off = 128; off > 0; off >>= 1) {
    if (tid < off) red[tid] += red[tid + off];
    __syncthreads();
  }
  float rs = rsqrtf(red[0] * (1.0f / (float)D_) + EPS_);
#pragma unroll
  for (int i = 0; i < 8; ++i) {
    int c = tid + i * 256;
    out[(size_t)row * D_ + c] = vals[i] * rs * w[c];
  }
}

// ---------------------------------------------------------------------------
// bf16 WMMA GEMM: C[M,N] = A[M,K] (bf16, row-major) x W[K,N] (fp32 -> bf16)
// Block 256 threads = 8 waves; tile BM=64, BN=256, BK=32.
// Wave (wm,wn) in 2x4 grid computes 32x64 = 2x4 WMMA frags (8 WMMA / K-step).
// A tile staged via async global->LDS b128 (ASYNCcnt); W tile staged through
// registers with fp32->bf16 conversion, stored TRANSPOSED so both A/B
// fragments load as 2x ds_load_b128. Double-buffered, W prefetched 2 ahead.
// ---------------------------------------------------------------------------
template <typename OutT>
__global__ __launch_bounds__(256)
void gemm_bf16_kernel(const __bf16* __restrict__ A, const float* __restrict__ W,
                      OutT* __restrict__ C, int M, int K, int N) {
  constexpr int BM = 64, BN = 256, BK = 32;
  __shared__ __bf16 sA[2][BM * BK];    // [row][k]      8 KB
  __shared__ __bf16 sBt[2][BN * BK];   // [col][k]     32 KB

  const int tid  = threadIdx.x;
  const int lane = tid & 31;
  const int wave = tid >> 5;
  const int wm   = wave >> 2;          // 0..1 -> row offset 32*wm
  const int wn   = wave & 3;           // 0..3 -> col offset 64*wn
  const int bm   = blockIdx.y * BM;
  const int bn   = blockIdx.x * BN;

  // Staging assignment: A -> one 16B chunk per thread; W -> 8 float4 chunks
  const int arow = tid >> 2;           // 0..63
  const int acol = (tid & 3) * 8;      // 0,8,16,24
  float4 wReg[8];

  auto asyncLoadA = [&](int k0, int buf) {
    async_copy_b128_to_lds(A + (size_t)(bm + arow) * K + k0 + acol,
                           &sA[buf][arow * BK + acol]);
  };
  auto loadGlobalW = [&](int k0) {
#pragma unroll
    for (int j = 0; j < 8; ++j) {
      int f = tid + j * 256;           // 0..2047
      int r = f >> 6;                  // k row 0..31
      int c4 = f & 63;                 // float4 col chunk
      wReg[j] = *(const float4*)(W + (size_t)(k0 + r) * N + bn + c4 * 4);
    }
    // prefetch W one more K-step ahead (global_prefetch_b8)
    if (k0 + BK < K) {
      int r = tid >> 6, c4 = tid & 63;
      __builtin_prefetch((const void*)(W + (size_t)(k0 + BK + r) * N + bn + c4 * 4), 0, 0);
    }
  };
  auto storeW = [&](int buf) {
#pragma unroll
    for (int j = 0; j < 8; ++j) {
      int f = tid + j * 256;
      int r = f >> 6;
      int c = (f & 63) * 4;
      sBt[buf][(c + 0) * BK + r] = (__bf16)wReg[j].x;
      sBt[buf][(c + 1) * BK + r] = (__bf16)wReg[j].y;
      sBt[buf][(c + 2) * BK + r] = (__bf16)wReg[j].z;
      sBt[buf][(c + 3) * BK + r] = (__bf16)wReg[j].w;
    }
  };

  v8f acc[2][4] = {};
  const int nsteps = K / BK;

  loadGlobalW(0);
  asyncLoadA(0, 0);
  storeW(0);
  wait_asynccnt0();
  __syncthreads();

  for (int step = 0; step < nsteps; ++step) {
    const int buf = step & 1;
    if (step + 1 < nsteps) {
      loadGlobalW((step + 1) * BK);
      asyncLoadA((step + 1) * BK, buf ^ 1);   // other buffer: safe, already idle
    }

    // A fragments: per-lane two contiguous 16B runs (documented 16x32 layout)
    v16bf af[2];
#pragma unroll
    for (int s = 0; s < 2; ++s) {
      int off = (wm * 32 + s * 16 + (lane & 15)) * BK + ((lane >> 4) << 3);
      v8bf lo = *(const v8bf*)(&sA[buf][off]);
      v8bf hi = *(const v8bf*)(&sA[buf][off + 16]);
      af[s] = __builtin_shufflevector(lo, hi, 0, 1, 2, 3, 4, 5, 6, 7,
                                      8, 9, 10, 11, 12, 13, 14, 15);
    }
    // B fragments: contiguous along k in transposed LDS
    v16bf bfr[4];
#pragma unroll
    for (int s = 0; s < 4; ++s) {
      int off = (wn * 64 + s * 16 + (lane & 15)) * BK + ((lane >> 4) << 3);
      v8bf lo = *(const v8bf*)(&sBt[buf][off]);
      v8bf hi = *(const v8bf*)(&sBt[buf][off + 16]);
      bfr[s] = __builtin_shufflevector(lo, hi, 0, 1, 2, 3, 4, 5, 6, 7,
                                       8, 9, 10, 11, 12, 13, 14, 15);
    }
#pragma unroll
    for (int si = 0; si < 2; ++si)
#pragma unroll
      for (int sj = 0; sj < 4; ++sj)
        acc[si][sj] = __builtin_amdgcn_wmma_f32_16x16x32_bf16(
            false, af[si], false, bfr[sj], (short)0, acc[si][sj], false, false);

    if (step + 1 < nsteps) {
      storeW(buf ^ 1);
      wait_asynccnt0();
      __syncthreads();
    }
  }

  // Store C: VGPR r -> (M = r + 8*hi, N = lane&15)
  const int hi   = lane >> 4;
  const int col0 = lane & 15;
#pragma unroll
  for (int si = 0; si < 2; ++si)
#pragma unroll
    for (int sj = 0; sj < 4; ++sj)
#pragma unroll
      for (int r = 0; r < 8; ++r) {
        int row = bm + wm * 32 + si * 16 + r + hi * 8;
        int col = bn + wn * 64 + sj * 16 + col0;
        C[(size_t)row * N + col] = (OutT)acc[si][sj][r];
      }
}

// ---------------------------------------------------------------------------
// RoPE: consume fp32 qkv[M, 3072], emit bf16 Q[M,HQ,HD], K[M,HKV,HD]
// ---------------------------------------------------------------------------
__global__ __launch_bounds__(256)
void rope_kernel(const float* __restrict__ qkv, const int* __restrict__ positions,
                 __bf16* __restrict__ Qb, __bf16* __restrict__ Kb) {
  size_t idx = (size_t)blockIdx.x * 256 + threadIdx.x;  // over M_*(HQ_+HKV_)*32
  int t   = (int)(idx / ((HQ_ + HKV_) * 32));
  int rem = (int)(idx % ((HQ_ + HKV_) * 32));
  int hh  = rem >> 5;
  int j   = rem & 31;

  float pos = (float)positions[t];
  float inv_freq = __expf(-(float)j * LN10000_OVER32);
  float ang = pos * inv_freq;
  float sn, cs;
  __sincosf(ang, &sn, &cs);

  if (hh < HQ_) {
    size_t base = (size_t)t * NQKV_ + (size_t)hh * HD_;
    float x1 = qkv[base + j];
    float x2 = qkv[base + j + 32];
    __bf16* dst = Qb + ((size_t)t * HQ_ + hh) * HD_;
    dst[j]      = (__bf16)(x1 * cs - x2 * sn);
    dst[j + 32] = (__bf16)(x2 * cs + x1 * sn);
  } else {
    int hk = hh - HQ_;
    size_t base = (size_t)t * NQKV_ + (size_t)HQ_ * HD_ + (size_t)hk * HD_;
    float x1 = qkv[base + j];
    float x2 = qkv[base + j + 32];
    __bf16* dst = Kb + ((size_t)t * HKV_ + hk) * HD_;
    dst[j]      = (__bf16)(x1 * cs - x2 * sn);
    dst[j + 32] = (__bf16)(x2 * cs + x1 * sn);
  }
}

// Copy V columns out of qkv (no RoPE), fp32 -> bf16
__global__ __launch_bounds__(256)
void vcopy_kernel(const float* __restrict__ qkv, __bf16* __restrict__ Vb) {
  size_t idx = (size_t)blockIdx.x * 256 + threadIdx.x;  // over M_*HKV_*HD_
  int t = (int)(idx >> 9);           // HKV_*HD_ = 512
  int c = (int)(idx & 511);
  Vb[idx] = (__bf16)qkv[(size_t)t * NQKV_ + (HQ_ + HKV_) * HD_ + c];
}

// ---------------------------------------------------------------------------
// Causal GQA attention, one wave per (b, head, 16-query tile).
// QK^T and P*V via WMMA bf16; raw scores live in LDS; softmax stats in regs;
// exp/normalize fused into the P-fragment load.
// ---------------------------------------------------------------------------
__global__ __launch_bounds__(32)
void attn_kernel(const __bf16* __restrict__ Qb, const __bf16* __restrict__ Kb,
                 const __bf16* __restrict__ Vb, __bf16* __restrict__ Ob) {
  __shared__ __bf16 probs[16 * S_];     // 32 KB raw scores

  int bid = blockIdx.x;                 // B_*HQ_*(S_/16) = 4096
  int qt = bid & 63;
  int h  = (bid >> 6) & (HQ_ - 1);
  int b  = bid >> 11;
  int hkv = h >> 2;                     // GROUP = 4
  int lane = threadIdx.x;
  int nkeys = (qt + 1) * 16;

  // Q fragments for the two HD chunks
  v16bf qf[2];
  {
    int qrow = qt * 16 + (lane & 15);
    size_t qbase = (((size_t)b * S_ + qrow) * HQ_ + h) * HD_;
#pragma unroll
    for (int c = 0; c < 2; ++c)
#pragma unroll
      for (int v = 0; v < 8; ++v) {
        int d = c * 32 + ((v >> 2) << 4) + ((lane >> 4) << 3) + ((v & 3) << 1);
        qf[c][2 * v]     = Qb[qbase + d];
        qf[c][2 * v + 1] = Qb[qbase + d + 1];
      }
  }

  // ---- Pass A: raw scores (scaled, causal-masked) into LDS ----
  for (int kt = 0; kt <= qt; ++kt) {
    v16bf kf[2];
    int key = kt * 16 + (lane & 15);
    size_t kbase = (((size_t)b * S_ + key) * HKV_ + hkv) * HD_;
#pragma unroll
    for (int c = 0; c < 2; ++c)
#pragma unroll
      for (int v = 0; v < 8; ++v) {
        int d = c * 32 + ((v >> 2) << 4) + ((lane >> 4) << 3) + ((v & 3) << 1);
        kf[c][2 * v]     = Kb[kbase + d];
        kf[c][2 * v + 1] = Kb[kbase + d + 1];
      }
    v8f sc = {};
    sc = __builtin_amdgcn_wmma_f32_16x16x32_bf16(false, qf[0], false, kf[0],
                                                 (short)0, sc, false, false);
    sc = __builtin_amdgcn_wmma_f32_16x16x32_bf16(false, qf[1], false, kf[1],
                                                 (short)0, sc, false, false);
    int n  = lane & 15;
    int mo = (lane >> 4) * 8;
#pragma unroll
    for (int r = 0; r < 8; ++r) {
      int m = r + mo;
      float val = sc[r] * SCALE_;
      if (kt == qt && n > m) val = -3.0e38f;
      probs[m * S_ + kt * 16 + n] = (__bf16)val;
    }
  }
  // pad tail to a 32-key multiple with -inf so PV chunks see exp() == 0
  if (nkeys & 31) {
#pragma unroll
    for (int r = 0; r < 8; ++r) {
      int m = lane & 15;
      int col = nkeys + ((lane >> 4) << 3) + r;
      probs[m * S_ + col] = (__bf16)(-3.0e38f);
    }
  }
  __syncthreads();

  // ---- Pass B: per-row max & sum(exp) kept in registers ----
  const int m    = lane & 15;
  const int half = lane >> 4;
  const __bf16* pr = probs + m * S_;
  float mx = -3.0e38f;
  for (int n = half; n < nkeys; n += 2) mx = fmaxf(mx, (float)pr[n]);
  mx = fmaxf(mx, __shfl_xor(mx, 16, 32));
  float sum = 0.0f;
  for (int n = half; n < nkeys; n += 2) sum += __expf((float)pr[n] - mx);
  sum += __shfl_xor(sum, 16, 32);
  const float inv = 1.0f / sum;
  __syncthreads();

  // ---- Pass C: O = P V over 32-key chunks; exp fused into P-frag load ----
  v8f oacc[4] = {};
  int nch = (nkeys + 31) >> 5;
  for (int kc = 0; kc < nch; ++kc) {
    v16bf pf;
#pragma unroll
    for (int v = 0; v < 8; ++v) {
      int kk = ((v >> 2) << 4) + ((lane >> 4) << 3) + ((v & 3) << 1);
      float p0 = __expf((float)pr[kc * 32 + kk] - mx) * inv;
      float p1 = __expf((float)pr[kc * 32 + kk + 1] - mx) * inv;
      pf[2 * v]     = (__bf16)p0;
      pf[2 * v + 1] = (__bf16)p1;
    }
#pragma unroll
    for (int nt = 0; nt < 4; ++nt) {
      v16bf vf;
      int d = nt * 16 + (lane & 15);
#pragma unroll
      for (int v = 0; v < 8; ++v) {
        int kk = ((v >> 2) << 4) + ((lane >> 4) << 3) + ((v & 3) << 1);
        int key = kc * 32 + kk;
        size_t vb = (((size_t)b * S_ + key) * HKV_ + hkv) * HD_ + d;
        vf[2 * v]     = Vb[vb];
        vf[2 * v + 1] = Vb[vb + (size_t)HKV_ * HD_];  // key+1
      }
      oacc[nt] = __builtin_amdgcn_wmma_f32_16x16x32_bf16(
          false, pf, false, vf, (short)0, oacc[nt], false, false);
    }
  }

  // Store O rows of [M, HQ*HD] (col = h*64 + d)
  int hi = lane >> 4;
  int n0 = lane & 15;
#pragma unroll
  for (int nt = 0; nt < 4; ++nt)
#pragma unroll
    for (int r = 0; r < 8; ++r) {
      int row = qt * 16 + r + hi * 8;
      size_t out = ((size_t)b * S_ + row) * (HQ_ * HD_) + h * HD_ + nt * 16 + n0;
      Ob[out] = (__bf16)oacc[nt][r];
    }
}

// ---------------------------------------------------------------------------
// SwiGLU: act[m,i] = silu(g)*u from bf16 gu[m, 2I]
// ---------------------------------------------------------------------------
__global__ __launch_bounds__(256)
void swiglu_kernel(const __bf16* __restrict__ gu, __bf16* __restrict__ act) {
  size_t idx = (size_t)blockIdx.x * 256 + threadIdx.x;  // over M_*I_
  int mrow = (int)(idx / I_);
  int i = (int)(idx % I_);
  float g = (float)gu[(size_t)mrow * (2 * I_) + i];
  float u = (float)gu[(size_t)mrow * (2 * I_) + I_ + i];
  float s = g / (1.0f + __expf(-g));
  act[idx] = (__bf16)(s * u);
}

// ---------------------------------------------------------------------------
// Host launcher
// ---------------------------------------------------------------------------
extern "C" void kernel_launch(void* const* d_in, const int* in_sizes, int n_in,
                              void* d_out, int out_size, void* d_ws, size_t ws_size,
                              hipStream_t stream) {
  (void)in_sizes; (void)n_in; (void)out_size; (void)ws_size;

  const int*   ids    = (const int*)d_in[0];
  const int*   pos    = (const int*)d_in[1];
  const float* embed  = (const float*)d_in[2];
  const float* w_qkv  = (const float*)d_in[3];
  const float* w_o    = (const float*)d_in[4];
  const float* w_gu   = (const float*)d_in[5];
  const float* w_down = (const float*)d_in[6];
  const float* ln1    = (const float*)d_in[7];
  const float* ln2    = (const float*)d_in[8];
  const float* normw  = (const float*)d_in[9];
  float* out = (float*)d_out;

  // Workspace carve-up (256B aligned)
  char* p = (char*)d_ws;
  auto alloc = [&](size_t bytes) -> void* {
    void* r = (void*)p;
    p += (bytes + 255) & ~(size_t)255;
    return r;
  };
  float*  h    = (float*)alloc((size_t)M_ * D_ * 4);
  float*  res  = (float*)alloc((size_t)M_ * D_ * 4);
  __bf16* hnb  = (__bf16*)alloc((size_t)M_ * D_ * 2);
  float*  qkv  = (float*)alloc((size_t)M_ * NQKV_ * 4);
  __bf16* Qb   = (__bf16*)alloc((size_t)M_ * HQ_ * HD_ * 2);
  __bf16* Kb   = (__bf16*)alloc((size_t)M_ * HKV_ * HD_ * 2);
  __bf16* Vb   = (__bf16*)alloc((size_t)M_ * HKV_ * HD_ * 2);
  __bf16* Ab   = (__bf16*)alloc((size_t)M_ * HQ_ * HD_ * 2);
  __bf16* gub  = (__bf16*)alloc((size_t)M_ * 2 * I_ * 2);
  __bf16* actb = (__bf16*)alloc((size_t)M_ * I_ * 2);

  embed_kernel<<<(M_ * D_) / 256, 256, 0, stream>>>(ids, embed, h, res);

  const int L = 4;
  for (int l = 0; l < L; ++l) {
    const float* wqkv_l = w_qkv  + (size_t)l * D_ * NQKV_;
    const float* wo_l   = w_o    + (size_t)l * (HQ_ * HD_) * D_;
    const float* wgu_l  = w_gu   + (size_t)l * D_ * (2 * I_);
    const float* wd_l   = w_down + (size_t)l * I_ * D_;
    const float* ln1_l  = ln1 + (size_t)l * D_;
    const float* ln2_l  = ln2 + (size_t)l * D_;

    addnorm_kernel<<<M_, 256, 0, stream>>>(h, res, ln1_l, hnb);

    gemm_bf16_kernel<float><<<dim3(NQKV_ / 256, M_ / 64), 256, 0, stream>>>(
        hnb, wqkv_l, qkv, M_, D_, NQKV_);

    rope_kernel<<<(M_ * (HQ_ + HKV_) * 32) / 256, 256, 0, stream>>>(qkv, pos, Qb, Kb);
    vcopy_kernel<<<(M_ * HKV_ * HD_) / 256, 256, 0, stream>>>(qkv, Vb);

    attn_kernel<<<B_ * HQ_ * (S_ / 16), 32, 0, stream>>>(Qb, Kb, Vb, Ab);

    gemm_bf16_kernel<float><<<dim3(D_ / 256, M_ / 64), 256, 0, stream>>>(
        Ab, wo_l, h, M_, HQ_ * HD_, D_);

    addnorm_kernel<<<M_, 256, 0, stream>>>(h, res, ln2_l, hnb);

    gemm_bf16_kernel<__bf16><<<dim3((2 * I_) / 256, M_ / 64), 256, 0, stream>>>(
        hnb, wgu_l, gub, M_, D_, 2 * I_);

    swiglu_kernel<<<(M_ * I_) / 256, 256, 0, stream>>>(gub, actb);

    gemm_bf16_kernel<float><<<dim3(D_ / 256, M_ / 64), 256, 0, stream>>>(
        actb, wd_l, h, M_, I_, D_);
  }

  finalnorm_kernel<<<M_, 256, 0, stream>>>(h, res, normw, out);
}
